// TriangleAttention_82308753260999
// MI455X (gfx1250) — matrix operand order repaired and checked by reading, not compiled, not run on hardware
//
#include <hip/hip_runtime.h>
#include <hip/hip_bf16.h>

typedef _Float16 half_t;
typedef __attribute__((ext_vector_type(16))) _Float16 v16h;
typedef __attribute__((ext_vector_type(8)))  _Float16 v8h;
typedef __attribute__((ext_vector_type(4)))  _Float16 v4h;
typedef __attribute__((ext_vector_type(8)))  float    v8f;

#define NRES 256
#define DPAIR 128
#define NHEAD 4
#define DHEAD 32
#define NPOS  (NRES*NRES)          // 65536 flattened (row, col) positions

__device__ __forceinline__ v16h cat8(v8h lo, v8h hi) {
    v16h r;
#pragma unroll
    for (int i = 0; i < 8; ++i) { r[i] = lo[i]; r[i + 8] = hi[i]; }
    return r;
}

__device__ __forceinline__ v8f wmma_f16(v16h a, v16h b, v8f c) {
    return __builtin_amdgcn_wmma_f32_16x16x32_f16(
        /*neg_a=*/false, a, /*neg_b=*/false, b,
        /*c_mod=*/(short)0, c, /*reuse_a=*/false, /*reuse_b=*/false);
}

__device__ __forceinline__ v8f zero8f() {
    v8f z = {0.f, 0.f, 0.f, 0.f, 0.f, 0.f, 0.f, 0.f};
    return z;
}

// fast hardware reciprocal (v_rcp_f32): plenty of accuracy next to f16 GEMMs
__device__ __forceinline__ float fast_rcp(float x) {
    return __builtin_amdgcn_rcpf(x);
}

// ---------------------------------------------------------------------------
// 0a) Convert a weight matrix f32 -> f16 (done once; removes per-tile cvts).
// ---------------------------------------------------------------------------
__global__ void __launch_bounds__(256)
cvtw_kernel(const float* __restrict__ src, half_t* __restrict__ dst, int n) {
    int t = blockIdx.x * blockDim.x + threadIdx.x;
    if (t < n) dst[t] = (half_t)src[t];
}

// 0b) wb_w [4,128] -> f16 padded to [16,128] with zero rows (for WMMA B tile).
__global__ void __launch_bounds__(256)
cvtpad_kernel(const float* __restrict__ src, half_t* __restrict__ dst) {
    int t = blockIdx.x * blockDim.x + threadIdx.x;
    if (t >= 16 * DPAIR) return;
    int r = t >> 7;
    dst[t] = (r < NHEAD) ? (half_t)src[t] : (half_t)0.f;
}

// ---------------------------------------------------------------------------
// 1) LayerNorm over D=128, one wave per (i,j) position, output f16.
// ---------------------------------------------------------------------------
__global__ void __launch_bounds__(256)
ln_kernel(const float* __restrict__ x, const float* __restrict__ g,
          const float* __restrict__ b, half_t* __restrict__ z) {
    int row  = (blockIdx.x * blockDim.x + threadIdx.x) >> 5;
    int lane = threadIdx.x & 31;
    if (row >= NPOS) return;
    const float* xr = x + (size_t)row * DPAIR;
    float4 v = *(const float4*)(xr + lane * 4);
    float s = v.x + v.y + v.z + v.w;
#pragma unroll
    for (int o = 16; o >= 1; o >>= 1) s += __shfl_xor(s, o, 32);
    float mu = s * (1.f / 128.f);
    float dx0 = v.x - mu, dx1 = v.y - mu, dx2 = v.z - mu, dx3 = v.w - mu;
    float q = dx0 * dx0 + dx1 * dx1 + dx2 * dx2 + dx3 * dx3;
#pragma unroll
    for (int o = 16; o >= 1; o >>= 1) q += __shfl_xor(q, o, 32);
    float r = rsqrtf(q * (1.f / 128.f) + 1e-5f);
    int d = lane * 4;
    v4h o;
    o[0] = (half_t)(dx0 * r * g[d + 0] + b[d + 0]);
    o[1] = (half_t)(dx1 * r * g[d + 1] + b[d + 1]);
    o[2] = (half_t)(dx2 * r * g[d + 2] + b[d + 2]);
    o[3] = (half_t)(dx3 * r * g[d + 3] + b[d + 3]);
    *(v4h*)(z + (size_t)row * DPAIR + d) = o;
}

// ---------------------------------------------------------------------------
// 2) Projection GEMM (compile-time specialized epilogue):
//    out16[r,n] = f16( post( scale * sum_d z16[r,d]*W[n,d] + bias[n] ) )
//    One 16x16 output tile per wave, K=128 as 4 x WMMA(K=32). W already f16.
// ---------------------------------------------------------------------------
template <bool HAS_BIAS, bool SIGMOID>
__global__ void __launch_bounds__(128)
proj_kernel(const half_t* __restrict__ Z, const half_t* __restrict__ W,
            const float* __restrict__ bias, half_t* __restrict__ out,
            float scale) {
    int wid  = (blockIdx.x * blockDim.x + threadIdx.x) >> 5;
    int lane = threadIdx.x & 31;
    int mtile = wid >> 3, ntile = wid & 7;
    int lm = lane & 15, hs = lane >> 4;

    const half_t* arow = Z + (size_t)(mtile * 16 + lm) * DPAIR;
    const half_t* brow = W + (size_t)(ntile * 16 + lm) * DPAIR;
    v8f c = zero8f();
#pragma unroll
    for (int kb = 0; kb < 4; ++kb) {
        int k0 = kb * 32;
        v16h A = cat8(*(const v8h*)(arow + k0 + hs * 8),
                      *(const v8h*)(arow + k0 + 16 + hs * 8));
        v16h B = cat8(*(const v8h*)(brow + k0 + hs * 8),
                      *(const v8h*)(brow + k0 + 16 + hs * 8));
        c = wmma_f16(A, B, c);
    }
    int n = ntile * 16 + lm;
    float bb = HAS_BIAS ? bias[n] : 0.f;
#pragma unroll
    for (int e = 0; e < 8; ++e) {
        float r = c[e] * scale + bb;
        if (SIGMOID) r = fast_rcp(1.f + __expf(-r));
        int m = mtile * 16 + e + 8 * hs;
        out[(size_t)m * DPAIR + n] = (half_t)r;
    }
}

// ---------------------------------------------------------------------------
// 3) Triangle bias, stored TRANSPOSED: triT[h, k, j] = z[(j,k)] . wb_w[h] + wb_b[h]
//    (transpose makes the attention-side bias read a contiguous 32B vector)
// ---------------------------------------------------------------------------
__global__ void __launch_bounds__(128)
tri_kernel(const half_t* __restrict__ Z, const half_t* __restrict__ wb16,
           const float* __restrict__ wbb, float* __restrict__ triT) {
    int mtile = (blockIdx.x * blockDim.x + threadIdx.x) >> 5;
    int lane = threadIdx.x & 31;
    int lm = lane & 15, hs = lane >> 4;
    bool bvalid = lm < NHEAD;

    const half_t* arow = Z + (size_t)(mtile * 16 + lm) * DPAIR;
    const half_t* brow = wb16 + (size_t)lm * DPAIR;  // rows >=4 are zero-padded
    v8f c = zero8f();
#pragma unroll
    for (int kb = 0; kb < 4; ++kb) {
        int k0 = kb * 32;
        v16h A = cat8(*(const v8h*)(arow + k0 + hs * 8),
                      *(const v8h*)(arow + k0 + 16 + hs * 8));
        v16h B = cat8(*(const v8h*)(brow + k0 + hs * 8),
                      *(const v8h*)(brow + k0 + 16 + hs * 8));
        c = wmma_f16(A, B, c);
    }
    if (bvalid) {
        float bb = wbb[lm];
#pragma unroll
        for (int e = 0; e < 8; ++e) {
            int m = mtile * 16 + e + 8 * hs;   // flattened (j,k): j=m>>8, k=m&255
            int j = m >> 8, k = m & 255;
            triT[(size_t)lm * NPOS + (size_t)k * NRES + j] = c[e] + bb;
        }
    }
}

// ---------------------------------------------------------------------------
// 4) Attention: one block per (i, h). 4 waves, each wave handles 4 j-tiles.
//    scores = q@k^T (+mask,+tri), softmax over k, y = P@v.
// ---------------------------------------------------------------------------
__global__ void __launch_bounds__(128)
attn_kernel(const half_t* __restrict__ Q, const half_t* __restrict__ K,
            const half_t* __restrict__ V, const float* __restrict__ mask,
            const float* __restrict__ triT, half_t* __restrict__ Y) {
    __shared__ __align__(16) half_t vT[DHEAD][NRES + 8];    // v transposed [c][k]
    __shared__ __align__(16) half_t pbuf[4][16][NRES + 8];  // per-wave P strip

    int i = blockIdx.x, h = blockIdx.y;
    int tid = threadIdx.x, lane = tid & 31, w = tid >> 5;
    int lm = lane & 15, hs = lane >> 4;

    // ---- stage V^T into LDS (shared by all 4 waves / 16 j-tiles) ----
    for (int r = tid; r < NRES; r += 128) {
        const half_t* src = V + (size_t)(i * NRES + r) * DPAIR + h * DHEAD;
#pragma unroll
        for (int cB = 0; cB < DHEAD; cB += 8) {
            v8h chunk = *(const v8h*)(src + cB);
#pragma unroll
            for (int t = 0; t < 8; ++t) vT[cB + t][r] = chunk[t];
        }
    }
    __syncthreads();

    const float* maskrow = mask + (size_t)i * NRES;

    for (int jt = 0; jt < 4; ++jt) {
        int jtile = w * 4 + jt;

        // q fragment: rows j, K = whole head dim (32)
        const half_t* qrow =
            Q + (size_t)(i * NRES + jtile * 16 + lm) * DPAIR + h * DHEAD;
        v16h Aq = cat8(*(const v8h*)(qrow + hs * 8),
                       *(const v8h*)(qrow + 16 + hs * 8));
        if (jt < 3) __builtin_prefetch(qrow + 16 * DPAIR, 0, 0);

        // ---- scores: 16 tiles of 16x16, one WMMA each (K=32) ----
        v8f sc[16];
#pragma unroll
        for (int kt = 0; kt < 16; ++kt) {
            const half_t* krow =
                K + (size_t)(i * NRES + kt * 16 + lm) * DPAIR + h * DHEAD + hs * 8;
            v16h Bk = cat8(*(const v8h*)krow, *(const v8h*)(krow + 16));
            sc[kt] = wmma_f16(Aq, Bk, zero8f());
        }

        // ---- bias add: mask over k (scalar) + triT[h][k][j] (32B vector) ----
#pragma unroll
        for (int kt = 0; kt < 16; ++kt) {
            int kp = kt * 16 + lm;
            float mb = 1.0e9f * (maskrow[kp] - 1.0f);
            v8f tv = *(const v8f*)(triT + (size_t)h * NPOS + (size_t)kp * NRES +
                                   jtile * 16 + 8 * hs);
#pragma unroll
            for (int e = 0; e < 8; ++e) sc[kt][e] += mb + tv[e];
        }

        // ---- softmax over k: in-lane over 16 tiles + shfl over 16-lane half ----
#pragma unroll
        for (int e = 0; e < 8; ++e) {
            float m = -3.0e38f;
#pragma unroll
            for (int kt = 0; kt < 16; ++kt) m = fmaxf(m, sc[kt][e]);
#pragma unroll
            for (int o = 8; o >= 1; o >>= 1) m = fmaxf(m, __shfl_xor(m, o, 32));
            float s = 0.f;
#pragma unroll
            for (int kt = 0; kt < 16; ++kt) {
                float p = __expf(sc[kt][e] - m);
                sc[kt][e] = p;
                s += p;
            }
#pragma unroll
            for (int o = 8; o >= 1; o >>= 1) s += __shfl_xor(s, o, 32);
            float inv = fast_rcp(s);
#pragma unroll
            for (int kt = 0; kt < 16; ++kt) sc[kt][e] *= inv;
        }

        // ---- reshape P (C-layout -> A-layout) via per-wave LDS strip ----
#pragma unroll
        for (int kt = 0; kt < 16; ++kt)
#pragma unroll
            for (int e = 0; e < 8; ++e)
                pbuf[w][e + 8 * hs][kt * 16 + lm] = (half_t)sc[kt][e];

        // ---- y = P @ v : 2 c-tiles x 8 K-chunks ----
#pragma unroll
        for (int cb = 0; cb < 2; ++cb) {
            v8f acc = zero8f();
#pragma unroll
            for (int kc = 0; kc < 8; ++kc) {
                const half_t* prow = &pbuf[w][lm][kc * 32 + hs * 8];
                v16h Ap = cat8(*(const v8h*)prow, *(const v8h*)(prow + 16));
                const half_t* vrow = &vT[cb * 16 + lm][kc * 32 + hs * 8];
                v16h Bv = cat8(*(const v8h*)vrow, *(const v8h*)(vrow + 16));
                acc = wmma_f16(Ap, Bv, acc);
            }
#pragma unroll
            for (int e = 0; e < 8; ++e) {
                int jg = jtile * 16 + e + 8 * hs;
                Y[(size_t)(i * NRES + jg) * DPAIR + h * DHEAD + cb * 16 + lm] =
                    (half_t)acc[e];
            }
        }
    }
}

// ---------------------------------------------------------------------------
// 5) Output: out[r,e] = sum_d (gate[r,d]*y[r,d]) * wo[e,d]   (f32 out)
// ---------------------------------------------------------------------------
__global__ void __launch_bounds__(128)
out_kernel(const half_t* __restrict__ Yh, const half_t* __restrict__ G,
           const half_t* __restrict__ Wo, float* __restrict__ out) {
    int wid  = (blockIdx.x * blockDim.x + threadIdx.x) >> 5;
    int lane = threadIdx.x & 31;
    int mtile = wid >> 3, ntile = wid & 7;
    int lm = lane & 15, hs = lane >> 4;

    const half_t* yrow = Yh + (size_t)(mtile * 16 + lm) * DPAIR;
    const half_t* grow = G  + (size_t)(mtile * 16 + lm) * DPAIR;
    const half_t* brow = Wo + (size_t)(ntile * 16 + lm) * DPAIR;
    v8f c = zero8f();
#pragma unroll
    for (int kb = 0; kb < 4; ++kb) {
        int k0 = kb * 32;
        v8h ylo = *(const v8h*)(yrow + k0 + hs * 8);
        v8h yhi = *(const v8h*)(yrow + k0 + 16 + hs * 8);
        v8h glo = *(const v8h*)(grow + k0 + hs * 8);
        v8h ghi = *(const v8h*)(grow + k0 + 16 + hs * 8);
        v16h A = cat8(ylo * glo, yhi * ghi);   // packed f16 multiplies
        v16h B = cat8(*(const v8h*)(brow + k0 + hs * 8),
                      *(const v8h*)(brow + k0 + 16 + hs * 8));
        c = wmma_f16(A, B, c);
    }
    int n = ntile * 16 + lm;
#pragma unroll
    for (int e = 0; e < 8; ++e) {
        int m = mtile * 16 + e + 8 * hs;
        out[(size_t)m * DPAIR + n] = c[e];
    }
}

// ---------------------------------------------------------------------------
extern "C" void kernel_launch(void* const* d_in, const int* in_sizes, int n_in,
                              void* d_out, int out_size, void* d_ws, size_t ws_size,
                              hipStream_t stream) {
    const float* pair = (const float*)d_in[0];
    const float* mask = (const float*)d_in[1];
    const float* ln_g = (const float*)d_in[2];
    const float* ln_b = (const float*)d_in[3];
    const float* wb_w = (const float*)d_in[4];
    const float* wb_b = (const float*)d_in[5];
    const float* wq   = (const float*)d_in[6];
    const float* wk   = (const float*)d_in[7];
    const float* wv   = (const float*)d_in[8];
    const float* wo   = (const float*)d_in[9];
    const float* wg_w = (const float*)d_in[10];
    const float* wg_b = (const float*)d_in[11];

    char* ws = (char*)d_ws;
    const size_t SZ = (size_t)NPOS * DPAIR * sizeof(half_t);  // 16 MB
    half_t* z16 = (half_t*)(ws);
    half_t* q16 = (half_t*)(ws + 1 * SZ);
    half_t* k16 = (half_t*)(ws + 2 * SZ);
    half_t* v16 = (half_t*)(ws + 3 * SZ);
    half_t* g16 = (half_t*)(ws + 4 * SZ);
    half_t* y16 = (half_t*)(ws + 5 * SZ);
    float*  tri = (float*)(ws + 6 * SZ);                      // 1 MB (transposed)
    char*   wbase = ws + 6 * SZ + (size_t)NHEAD * NPOS * sizeof(float);
    const size_t WSZ = (size_t)DPAIR * DPAIR * sizeof(half_t);  // 32 KB
    half_t* wq16 = (half_t*)(wbase + 0 * WSZ);
    half_t* wk16 = (half_t*)(wbase + 1 * WSZ);
    half_t* wv16 = (half_t*)(wbase + 2 * WSZ);
    half_t* wg16 = (half_t*)(wbase + 3 * WSZ);
    half_t* wo16 = (half_t*)(wbase + 4 * WSZ);
    half_t* wb16 = (half_t*)(wbase + 5 * WSZ);  // padded 16x128
    float*  outp = (float*)d_out;

    // ---- one-time weight conversions (tiny) ----
    const int NW = DPAIR * DPAIR;
    cvtw_kernel<<<NW / 256, 256, 0, stream>>>(wq, wq16, NW);
    cvtw_kernel<<<NW / 256, 256, 0, stream>>>(wk, wk16, NW);
    cvtw_kernel<<<NW / 256, 256, 0, stream>>>(wv, wv16, NW);
    cvtw_kernel<<<NW / 256, 256, 0, stream>>>(wg_w, wg16, NW);
    cvtw_kernel<<<NW / 256, 256, 0, stream>>>(wo, wo16, NW);
    cvtpad_kernel<<<8, 256, 0, stream>>>(wb_w, wb16);

    // 65536 rows, one wave each (8 per 256-thread block)
    ln_kernel<<<NPOS / 8, 256, 0, stream>>>(pair, ln_g, ln_b, z16);

    // 4096 m-tiles x 8 n-tiles, 4 wave-tiles per 128-thread block
    const int gproj = (NPOS / 16) * (DPAIR / 16) / 4;  // 8192
    proj_kernel<false, false><<<gproj, 128, 0, stream>>>(
        z16, wq16, nullptr, q16, 0.17677669529663687f);
    proj_kernel<false, false><<<gproj, 128, 0, stream>>>(z16, wk16, nullptr, k16,
                                                         1.0f);
    proj_kernel<false, false><<<gproj, 128, 0, stream>>>(z16, wv16, nullptr, v16,
                                                         1.0f);
    proj_kernel<true, true><<<gproj, 128, 0, stream>>>(z16, wg16, wg_b, g16,
                                                       1.0f);

    tri_kernel<<<(NPOS / 16) / 4, 128, 0, stream>>>(z16, wb16, wb_b, tri);

    attn_kernel<<<dim3(NRES, NHEAD), 128, 0, stream>>>(q16, k16, v16, mask, tri,
                                                       y16);

    out_kernel<<<gproj, 128, 0, stream>>>(y16, g16, wo16, outp);
}
